// CustomMultiheadAttention_1030792151430
// MI455X (gfx1250) — matrix operand order repaired and checked by reading, not compiled, not run on hardware
//
#include <hip/hip_runtime.h>

#define B_   4
#define N_   512
#define DIN  256
#define E_   64
#define H_   4

typedef __attribute__((ext_vector_type(16))) __bf16 v16bf;
typedef __attribute__((ext_vector_type(8)))  __bf16 v8bf;
typedef __attribute__((ext_vector_type(8)))  float  v8f;

__device__ __forceinline__ v8f wmma_bf16(v16bf a, v16bf b, v8f c) {
    // D = A(16x32 bf16) * B(32x16 bf16) + C(16x16 f32)
    return __builtin_amdgcn_wmma_f32_16x16x32_bf16(
        /*neg_a=*/false, a, /*neg_b=*/false, b,
        /*c_mod=*/(short)0, c, /*reuse_a=*/false, /*reuse_b=*/false);
}

__device__ __forceinline__ v16bf pack16(v8bf a, v8bf b) {
    return __builtin_shufflevector(a, b, 0, 1, 2, 3, 4, 5, 6, 7,
                                         8, 9, 10, 11, 12, 13, 14, 15);
}

// ---------------------------------------------------------------- conversions
// x -> bf16 row-major; W -> bf16 TRANSPOSED per head: [H, E, DIN]
__global__ void cvt_kernel(const float* __restrict__ x,
                           const float* __restrict__ wq,
                           const float* __restrict__ wk,
                           const float* __restrict__ wv,
                           __bf16* __restrict__ xb,
                           __bf16* __restrict__ wqt,
                           __bf16* __restrict__ wkt,
                           __bf16* __restrict__ wvt) {
    const int nx = B_ * N_ * DIN;     // 524288
    const int nw = H_ * DIN * E_;     // 65536
    int i = blockIdx.x * blockDim.x + threadIdx.x;
    if (i < nx) {
        xb[i] = (__bf16)x[i];
    } else if (i < nx + 3 * nw) {
        const int which = (i - nx) / nw;
        const int j = (i - nx) % nw;
        const int h = j / (DIN * E_);
        const int r = j % (DIN * E_);
        const int d = r / E_;
        const int e = r % E_;
        const int t = (h * E_ + e) * DIN + d;
        if (which == 0)      wqt[t] = (__bf16)wq[j];
        else if (which == 1) wkt[t] = (__bf16)wk[j];
        else                 wvt[t] = (__bf16)wv[j];
    }
}

// ---------------------------------------------------------------- projection
// out = x @ W + bias.  Wt layout [H, E, DIN] so B fragments are contiguous.
// store_t == 0: out[b,h,n,e] (row major);  store_t == 1: out[b,h,e,n] (transposed)
// grid: (B_*H_, N_/32), block: 64 (2 waves, 16 rows each)
__global__ __launch_bounds__(64)
void proj_kernel(const __bf16* __restrict__ xb,
                 const __bf16* __restrict__ Wt,    // [H, E, DIN]
                 const float*  __restrict__ bias,  // [H, E]
                 __bf16* __restrict__ outb, int store_t)
{
    const int bh = blockIdx.x;
    const int b = bh / H_, h = bh % H_;
    const int wave = threadIdx.x >> 5;
    const int lane = threadIdx.x & 31;
    const int lo = lane & 15, hi = lane >> 4;
    const int rowbase = blockIdx.y * 32 + wave * 16;

    const __bf16* X = xb + (size_t)(b * N_ + rowbase) * DIN;
    const __bf16* W = Wt + (size_t)h * E_ * DIN;

    v8f acc[4] = {};
    for (int ks = 0; ks < DIN / 32; ++ks) {
        const int kb = ks * 32;
        const __bf16* ap = X + lo * DIN + kb + 8 * hi;
        v16bf a = pack16(*(const v8bf*)ap, *(const v8bf*)(ap + 16));
#pragma unroll
        for (int ct = 0; ct < 4; ++ct) {
            const int e = ct * 16 + lo;
            const __bf16* wp = W + (size_t)e * DIN + kb + 8 * hi;
            v16bf bb = pack16(*(const v8bf*)wp, *(const v8bf*)(wp + 16));
            acc[ct] = wmma_bf16(a, bb, acc[ct]);
        }
    }
#pragma unroll
    for (int ct = 0; ct < 4; ++ct) {
        const int e = ct * 16 + lo;
        const float bs = bias[h * E_ + e];
#pragma unroll
        for (int r = 0; r < 8; ++r) {
            const int row = rowbase + r + 8 * hi;
            const __bf16 v = (__bf16)(acc[ct][r] + bs);
            if (store_t)
                outb[((size_t)(b * H_ + h) * E_ + e) * N_ + row] = v;
            else
                outb[((size_t)(b * H_ + h) * N_ + row) * E_ + e] = v;
        }
    }
}

// ---------------------------------------------------------------- row norms
// grid: B_*H_*N_, block: 64. qsq/ksq = squared row norms; Qnb/Knb = unit rows.
__global__ __launch_bounds__(64)
void norms_kernel(const __bf16* __restrict__ Qb, const __bf16* __restrict__ Kb,
                  __bf16* __restrict__ Qnb, __bf16* __restrict__ Knb,
                  float* __restrict__ qsq, float* __restrict__ ksq)
{
    __shared__ float sq[64], sk[64];
    const int row = blockIdx.x;
    const int e = threadIdx.x;
    const float q = (float)Qb[(size_t)row * E_ + e];
    const float k = (float)Kb[(size_t)row * E_ + e];
    sq[e] = q * q; sk[e] = k * k;
    __syncthreads();
    for (int s = 32; s > 0; s >>= 1) {
        if (e < s) { sq[e] += sq[e + s]; sk[e] += sk[e + s]; }
        __syncthreads();
    }
    const float qs = sq[0], ks = sk[0];
    if (e == 0) { qsq[row] = qs; ksq[row] = ks; }
    Qnb[(size_t)row * E_ + e] = (__bf16)(q * rsqrtf(fmaxf(qs, 1e-20f)));
    Knb[(size_t)row * E_ + e] = (__bf16)(k * rsqrtf(fmaxf(ks, 1e-20f)));
}

// -------------------------------------------------- softmax + P@V (shared)
// lds_logits: [16][N_] f32 for this block's 16 query rows.
// Vt: [B,H,E,N] transposed, so B fragments are contiguous along m.
__device__ __forceinline__
void softmax_pv_store(float* lds_logits, float* s_max, float* s_rs,
                      const __bf16* __restrict__ Vt,
                      float* __restrict__ out,
                      int b, int h, int n0, int lane, int lo, int hi)
{
    // per-row max and sum(exp) via wave32 shuffles
    for (int r = 0; r < 16; ++r) {
        float mx = -1e30f;
        for (int i = 0; i < N_ / 32; ++i)
            mx = fmaxf(mx, lds_logits[r * N_ + lane + 32 * i]);
#pragma unroll
        for (int s = 16; s > 0; s >>= 1) mx = fmaxf(mx, __shfl_xor(mx, s, 32));
        float sum = 0.f;
        for (int i = 0; i < N_ / 32; ++i)
            sum += __expf(lds_logits[r * N_ + lane + 32 * i] - mx);
#pragma unroll
        for (int s = 16; s > 0; s >>= 1) sum += __shfl_xor(sum, s, 32);
        if (lane == 0) { s_max[r] = mx; s_rs[r] = 1.0f / sum; }
    }
    __syncthreads();

    const __bf16* Vh = Vt + (size_t)(b * H_ + h) * E_ * N_;
    __builtin_prefetch(Vh + (size_t)lane * N_, 0, 1);

    const float rmax = s_max[lo];
    const float rrs  = s_rs[lo];
    v8f acc[4] = {};
    for (int ks = 0; ks < N_ / 32; ++ks) {
        v16bf a;
#pragma unroll
        for (int j = 0; j < 16; ++j) {
            const int m = ks * 32 + ((j < 8) ? j : j + 8) + 8 * hi;
            const float p = __expf(lds_logits[lo * N_ + m] - rmax) * rrs;
            a[j] = (__bf16)p;
        }
#pragma unroll
        for (int nt = 0; nt < 4; ++nt) {
            const int e = nt * 16 + lo;
            const __bf16* vp = Vh + (size_t)e * N_ + ks * 32 + 8 * hi;
            v16bf bb = pack16(*(const v8bf*)vp, *(const v8bf*)(vp + 16));
            acc[nt] = wmma_bf16(a, bb, acc[nt]);
        }
    }
#pragma unroll
    for (int nt = 0; nt < 4; ++nt) {
#pragma unroll
        for (int r = 0; r < 8; ++r) {
            const int row = n0 + r + 8 * hi;
            out[((size_t)b * N_ + row) * (H_ * E_) + h * E_ + nt * 16 + lo] = acc[nt][r];
        }
    }
}

// ------------------------------------------- heads 0 (sdp), 1 (cosine), 3 (l2)
// grid: (B_, N_/16), block: 32 (one wave owns 16 query rows)
__global__ __launch_bounds__(32)
void attn_dot_kernel(const __bf16* __restrict__ Qsrc, const __bf16* __restrict__ Ksrc,
                     const __bf16* __restrict__ Vt,
                     const float* __restrict__ qsq, const float* __restrict__ ksq,
                     float* __restrict__ out, int h, float scale, int use_l2)
{
    __shared__ float lds_logits[16 * N_];   // 32 KB
    __shared__ float s_max[16], s_rs[16];

    const int b = blockIdx.x, n0 = blockIdx.y * 16;
    const int lane = threadIdx.x;
    const int lo = lane & 15, hi = lane >> 4;
    const size_t base = (size_t)(b * H_ + h) * N_;

    const __bf16* Q = Qsrc + (base + n0) * E_;
    const __bf16* K = Ksrc + base * E_;

    // A fragments: Q rows, E split into two K=32 steps
    v16bf aq[2];
#pragma unroll
    for (int ks = 0; ks < 2; ++ks) {
        const __bf16* p = Q + lo * E_ + ks * 32 + 8 * hi;
        aq[ks] = pack16(*(const v8bf*)p, *(const v8bf*)(p + 16));
    }
    float qs[8];
#pragma unroll
    for (int r = 0; r < 8; ++r) qs[r] = qsq[base + n0 + r + 8 * hi];

    for (int mt = 0; mt < N_ / 16; ++mt) {
        v8f acc = {};
#pragma unroll
        for (int ks = 0; ks < 2; ++ks) {
            // B fragment: element (k=e, n=m); two 8-bf16 runs along e (table order)
            const __bf16* p = K + (size_t)(mt * 16 + lo) * E_ + ks * 32 + 8 * hi;
            v16bf bb = pack16(*(const v8bf*)p, *(const v8bf*)(p + 16));
            acc = wmma_bf16(aq[ks], bb, acc);
        }
        const float kcol = ksq[base + mt * 16 + lo];
#pragma unroll
        for (int r = 0; r < 8; ++r) {
            float lg;
            if (use_l2) {
                const float d2 = fmaxf(qs[r] + kcol - 2.f * acc[r], 0.f);
                lg = -sqrtf(d2);
            } else {
                lg = acc[r] * scale;
            }
            lds_logits[(r + 8 * hi) * N_ + mt * 16 + lo] = lg;
        }
    }
    __syncthreads();
    softmax_pv_store(lds_logits, s_max, s_rs, Vt, out, b, h, n0, lane, lo, hi);
}

// -------------------------------------------------------------- head 2 (L1)
// grid: (B_, N_/16), block: 32.  K chunks staged to LDS with the gfx1250
// async-to-LDS data path (global_load_async_to_lds_b128 + s_wait_asynccnt).
__global__ __launch_bounds__(32)
void attn_l1_kernel(const __bf16* __restrict__ Qb, const __bf16* __restrict__ Kb,
                    const __bf16* __restrict__ Vt, float* __restrict__ out, int h)
{
    __shared__ float  lds_logits[16 * N_];  // 32 KB
    __shared__ float  s_max[16], s_rs[16];
    __shared__ float  qt[16 * E_];          // 4 KB
    __shared__ __bf16 kt[32 * E_];          // 4 KB (raw bf16 chunk of K)

    const int b = blockIdx.x, n0 = blockIdx.y * 16;
    const int lane = threadIdx.x;
    const int lo = lane & 15, hi = lane >> 4;
    const size_t base = (size_t)(b * H_ + h) * N_;

    for (int i = lane; i < 16 * E_; i += 32)
        qt[i] = (float)Qb[(base + n0) * E_ + i];
    __syncthreads();

    const unsigned kt_lds = (unsigned)(size_t)(&kt[0]);
    for (int mc = 0; mc < N_ / 32; ++mc) {
        // async copy 32x64 bf16 (4 KB) of K into LDS: 8 x b128 per lane
        const char* src = (const char*)(Kb + (base + mc * 32) * E_);
#pragma unroll
        for (int i = 0; i < 8; ++i) {
            const unsigned byteoff = (unsigned)(i * 32 + lane) * 16u;
            const unsigned ldsa = kt_lds + byteoff;
            const void* gp = src + byteoff;
            asm volatile("global_load_async_to_lds_b128 %0, %1, off"
                         :: "v"(ldsa), "v"(gp) : "memory");
        }
        asm volatile("s_wait_asynccnt 0x0" ::: "memory");
        __syncthreads();
        for (int r = 0; r < 16; ++r) {
            float d = 0.f;
#pragma unroll 8
            for (int e = 0; e < E_; ++e)
                d += fabsf(qt[r * E_ + e] - (float)kt[lane * E_ + e]);
            lds_logits[r * N_ + mc * 32 + lane] = -d;
        }
        __syncthreads();
    }
    softmax_pv_store(lds_logits, s_max, s_rs, Vt, out, b, h, n0, lane, lo, hi);
}

// ------------------------------------------------------------------- launch
extern "C" void kernel_launch(void* const* d_in, const int* in_sizes, int n_in,
                              void* d_out, int out_size, void* d_ws, size_t ws_size,
                              hipStream_t stream)
{
    const float* x  = (const float*)d_in[0];
    const float* Wq = (const float*)d_in[1];
    const float* bq = (const float*)d_in[2];
    const float* Wk = (const float*)d_in[3];
    const float* bk = (const float*)d_in[4];
    const float* Wv = (const float*)d_in[5];
    const float* bv = (const float*)d_in[6];
    float* out = (float*)d_out;

    char* ws = (char*)d_ws;
    size_t off = 0;
    auto alloc = [&](size_t bytes) -> void* {
        void* p = ws + off;
        off = (off + bytes + 255) & ~(size_t)255;
        return p;
    };
    const size_t nQKV = (size_t)B_ * H_ * N_ * E_;  // 524288
    __bf16* xb  = (__bf16*)alloc((size_t)B_ * N_ * DIN * 2);
    __bf16* wqt = (__bf16*)alloc((size_t)H_ * DIN * E_ * 2);
    __bf16* wkt = (__bf16*)alloc((size_t)H_ * DIN * E_ * 2);
    __bf16* wvt = (__bf16*)alloc((size_t)H_ * DIN * E_ * 2);
    __bf16* Qb  = (__bf16*)alloc(nQKV * 2);   // [B,H,N,E]
    __bf16* Kb  = (__bf16*)alloc(nQKV * 2);   // [B,H,N,E]
    __bf16* Vt  = (__bf16*)alloc(nQKV * 2);   // [B,H,E,N] transposed
    __bf16* Qnb = (__bf16*)alloc(nQKV * 2);   // [B,H,N,E] normalized
    __bf16* Knb = (__bf16*)alloc(nQKV * 2);   // [B,H,N,E] normalized
    float*  qsq = (float*)alloc((size_t)B_ * H_ * N_ * 4);
    float*  ksq = (float*)alloc((size_t)B_ * H_ * N_ * 4);

    // 1) f32 -> bf16 conversions (W transposed on the fly)
    const int total = B_ * N_ * DIN + 3 * H_ * DIN * E_;
    cvt_kernel<<<(total + 255) / 256, 256, 0, stream>>>(x, Wq, Wk, Wv, xb, wqt, wkt, wvt);

    // 2) QKV projections (WMMA); V written transposed for the P@V B operand
    dim3 pg(B_ * H_, N_ / 32);
    proj_kernel<<<pg, 64, 0, stream>>>(xb, wqt, bq, Qb, 0);
    proj_kernel<<<pg, 64, 0, stream>>>(xb, wkt, bk, Kb, 0);
    proj_kernel<<<pg, 64, 0, stream>>>(xb, wvt, bv, Vt, 1);

    // 3) row norms + normalized copies (for cosine and L2 heads)
    norms_kernel<<<B_ * H_ * N_, 64, 0, stream>>>(Qb, Kb, Qnb, Knb, qsq, ksq);

    // 4) four attention heads
    dim3 ag(B_, N_ / 16);
    attn_dot_kernel<<<ag, 32, 0, stream>>>(Qb,  Kb,  Vt, qsq, ksq, out, 0, 0.125f, 0); // sdp
    attn_dot_kernel<<<ag, 32, 0, stream>>>(Qnb, Knb, Vt, qsq, ksq, out, 1, 1.0f,   0); // cosine
    attn_l1_kernel <<<ag, 32, 0, stream>>>(Qb,  Kb,  Vt, out, 2);                      // L1
    attn_dot_kernel<<<ag, 32, 0, stream>>>(Qb,  Kb,  Vt, qsq, ksq, out, 3, 1.0f,   1); // L2
}
